// MultiHeadLatentAttention_35021163331633
// MI455X (gfx1250) — matrix operand order repaired and checked by reading, not compiled, not run on hardware
//
#include <hip/hip_runtime.h>
#include <hip/hip_bf16.h>

typedef __bf16 bf16;
typedef __attribute__((ext_vector_type(16))) __bf16 bf16x16;
typedef __attribute__((ext_vector_type(8)))  __bf16 bf16x8;
typedef __attribute__((ext_vector_type(8)))  float  f32x8;
typedef __attribute__((ext_vector_type(4)))  unsigned int u32x4;
typedef __attribute__((ext_vector_type(8)))  int i32x8;
typedef __attribute__((ext_vector_type(4)))  int i32x4;

#define B_    2
#define T_    2048
#define HID_  2048
#define NH_   16
#define NKV_  8
#define NOPE_ 128
#define ROPE_ 64
#define HD_   192
#define VD_   128
#define KVR_  512
#define QR_   1536
#define ROWS_ (B_ * T_)
#define EPS_  1e-6f
// ln(10000)/32
#define LOG_THETA_OVER_HALF 0.28782313662425572f

__device__ __forceinline__ bf16x16 mk_frag(bf16x8 lo, bf16x8 hi) {
  return __builtin_shufflevector(lo, hi, 0,1,2,3,4,5,6,7,8,9,10,11,12,13,14,15);
}
__device__ __forceinline__ f32x8 wmma_bf16(bf16x16 a, bf16x16 b, f32x8 c) {
  return __builtin_amdgcn_wmma_f32_16x16x32_bf16(false, a, false, b, (short)0, c,
                                                 false, false);
}
__device__ __forceinline__ float block_sum(float v) {
#pragma unroll
  for (int off = 16; off; off >>= 1) v += __shfl_xor(v, off, 32);
  __shared__ float red[8];
  if ((threadIdx.x & 31) == 0) red[threadIdx.x >> 5] = v;
  __syncthreads();
  float s = 0.f;
#pragma unroll
  for (int i = 0; i < 8; ++i) s += red[i];
  __syncthreads();
  return s;
}

// ---------------------------------------------------------------------------
// TDM: DMA a 128-row x 32-col bf16 tile (row stride K elems) from global into
// LDS with 16B padding after every 64B row -> padded row stride 80B (40 elems),
// matching the conflict-free fragment-read layout.
//   pad_interval = 3 (pad after 16 DWORDs = 64B)
//   pad_amount   = 3 (insert 4 DWORDs = 16B)
// D# packing per CDNA5 ISA ch.8 (group0: count/lds/global/type, group1: dims).
// This toolchain's builtin is the 6-arg form:
//   (u32x4 g0, i32x8 g1, i32x4 g2, i32x4 g3, i32x8 extra, i32 cpol)
// ---------------------------------------------------------------------------
__device__ __forceinline__ void tdm_load_tile_128x32(const bf16* g, bf16* lds,
                                                     int K) {
  const unsigned long long ga = (unsigned long long)(uintptr_t)g;
  const unsigned lds_off = (unsigned)(uintptr_t)lds;  // low 32b = LDS offset
  u32x4 g0;
  g0[0] = 1u;                                   // count=1, user load
  g0[1] = lds_off;                              // lds_addr
  g0[2] = (unsigned)(ga & 0xFFFFFFFFu);         // global_addr[31:0]
  g0[3] = (unsigned)((ga >> 32) & 0x1FFFFFFu)   // global_addr[56:32]
          | (2u << 30);                         // type = 2 ("image")
  i32x8 g1;
  g1[0] = (1 << 16)        // data_size = 1 -> 2 bytes
        | (1 << 20)        // pad_enable
        | (3 << 22)        // pad_interval: 16 DWORDs
        | (3 << 25);       // pad_amount:   4 DWORDs
  g1[1] = (int)((unsigned)(K & 0xFFFF) << 16);            // tensor_dim0[15:0]
  g1[2] = (int)(((unsigned)K >> 16) & 0xFFFFu)            // tensor_dim0[31:16]
        | (128 << 16);                                    // tensor_dim1[15:0]
  g1[3] = (32 << 16);                                     // tile_dim0 = 32
  g1[4] = 128;                                            // tile_dim1 = 128
  g1[5] = K;                                              // dim0_stride[31:0]
  g1[6] = 0;                                              // stride hi / dim1_stride lo
  g1[7] = 0;
  const i32x4 z4 = {0, 0, 0, 0};                // 2-D tile: groups 2/3 unused
  const i32x8 z8 = {0, 0, 0, 0, 0, 0, 0, 0};
  __builtin_amdgcn_tensor_load_to_lds(g0, g1, z4, z4, z8, 0);
}

// ---------------------------------------------------------------------------
// C[M,N] f32 = A[M,K] bf16 @ Bt[N,K] bf16   (Bt is the transposed weight)
// Block: 256 thr = 8 waves, 128x128 tile, K step 32. Wave = 32(M) x 64(N).
// Double-buffered: wave 0 issues TDM loads for tile kt+1 while all waves
// compute tile kt; sync on TENSORcnt + workgroup barrier.
// ---------------------------------------------------------------------------
__global__ __launch_bounds__(256) void mla_gemm_bf16(
    const bf16* __restrict__ A, const bf16* __restrict__ Bt,
    float* __restrict__ C, int M, int N, int K) {
  __shared__ alignas(16) bf16 Asm[2][128 * 40];
  __shared__ alignas(16) bf16 Bsm[2][128 * 40];
  const int tid  = threadIdx.x;
  const int lane = tid & 31;
  const int wave = tid >> 5;
  const int hlf  = lane >> 4;
  const int ln   = lane & 15;
  const int m0   = blockIdx.y * 128;
  const int n0   = blockIdx.x * 128;
  const int wm   = (wave & 3) * 32;
  const int wn   = (wave >> 2) * 64;
  f32x8 acc[2][4] = {};
  const int kTiles = K >> 5;

  if (wave == 0) {
    tdm_load_tile_128x32(A + (size_t)m0 * K, &Asm[0][0], K);
    tdm_load_tile_128x32(Bt + (size_t)n0 * K, &Bsm[0][0], K);
  }
  int buf = 0;
  for (int kt = 0; kt < kTiles; ++kt) {
    if (wave == 0 && kt + 1 < kTiles) {
      const int k1 = (kt + 1) << 5;
      tdm_load_tile_128x32(A + (size_t)m0 * K + k1, &Asm[buf ^ 1][0], K);
      tdm_load_tile_128x32(Bt + (size_t)n0 * K + k1, &Bsm[buf ^ 1][0], K);
      __builtin_amdgcn_s_wait_tensorcnt(2);  // current tile's 2 DMAs done
    } else {
      __builtin_amdgcn_s_wait_tensorcnt(0);
    }
    __syncthreads();  // publish LDS tile to all waves
    bf16x16 aF[2], bF[4];
#pragma unroll
    for (int i = 0; i < 2; ++i) {  // A layout: chunks at 8h and 16+8h
      const int r = wm + i * 16 + ln;
      aF[i] = mk_frag(*(const bf16x8*)(&Asm[buf][r * 40 + hlf * 8]),
                      *(const bf16x8*)(&Asm[buf][r * 40 + 16 + hlf * 8]));
    }
#pragma unroll
    for (int j = 0; j < 4; ++j) {  // B layout: chunks at 8h and 16+8h
      const int r = wn + j * 16 + ln;
      bF[j] = mk_frag(*(const bf16x8*)(&Bsm[buf][r * 40 + hlf * 8]),
                      *(const bf16x8*)(&Bsm[buf][r * 40 + 16 + hlf * 8]));
    }
#pragma unroll
    for (int i = 0; i < 2; ++i)
#pragma unroll
      for (int j = 0; j < 4; ++j) acc[i][j] = wmma_bf16(aF[i], bF[j], acc[i][j]);
    __syncthreads();  // all waves done reading buf before it is re-filled
    buf ^= 1;
  }
#pragma unroll
  for (int i = 0; i < 2; ++i)
#pragma unroll
    for (int j = 0; j < 4; ++j)
#pragma unroll
      for (int r = 0; r < 8; ++r) {
        const int m = m0 + wm + i * 16 + r + hlf * 8;
        const int n = n0 + wn + j * 16 + ln;
        C[(size_t)m * N + n] = acc[i][j][r];
      }
}

// ---------------------------------------------------------------------------
// Flash attention: wave = one 16-query tile of one (b,h); key blocks of 32.
// Q[B,NH,T,HD] bf16, Kf[B,NKV,T,HD] bf16, Vt[B,NKV,VD,T] bf16 -> Oa[B,T,NH*VD]
// ---------------------------------------------------------------------------
__global__ __launch_bounds__(256) void mla_attention(
    const bf16* __restrict__ Q, const bf16* __restrict__ Kf,
    const bf16* __restrict__ Vt, bf16* __restrict__ Oa) {
  __shared__ alignas(16) bf16 Psm[8][16 * 40];
  const int tid  = threadIdx.x;
  const int lane = tid & 31;
  const int wave = tid >> 5;
  const int hlf  = lane >> 4;
  const int ln   = lane & 15;
  const int h    = blockIdx.y;
  const int b    = blockIdx.z;
  const int kvh  = h >> 1;  // GROUPS = 2
  const int q0   = blockIdx.x * 128 + wave * 16;
  const float scale = 0.07216878364870322f;  // 1/sqrt(192)

  const bf16* qBase = Q + (((size_t)b * NH_ + h) * T_ + q0) * HD_;
  bf16x16 qF[6];
#pragma unroll
  for (int kf = 0; kf < 6; ++kf) {
    const bf16* p = qBase + (size_t)ln * HD_ + kf * 32;
    qF[kf] = mk_frag(*(const bf16x8*)(p + hlf * 8),
                     *(const bf16x8*)(p + 16 + hlf * 8));
  }

  float mrow[8], lrow[8];
  f32x8 o[8] = {};
#pragma unroll
  for (int r = 0; r < 8; ++r) { mrow[r] = -3.0e38f; lrow[r] = 0.f; }

  const bf16* kHead = Kf + ((size_t)b * NKV_ + kvh) * T_ * HD_;
  const bf16* vHead = Vt + ((size_t)b * NKV_ + kvh) * VD_ * T_;
  bf16* pw = &Psm[wave][0];

  for (int kb = 0; kb < T_ / 32; ++kb) {
    f32x8 s0 = {}, s1 = {};
    const bf16* k0p = kHead + (size_t)(kb * 32 + ln) * HD_;
    const bf16* k1p = kHead + (size_t)(kb * 32 + 16 + ln) * HD_;
#pragma unroll
    for (int kf = 0; kf < 6; ++kf) {
      bf16x16 b0 = mk_frag(*(const bf16x8*)(k0p + kf * 32 + hlf * 8),
                           *(const bf16x8*)(k0p + kf * 32 + 16 + hlf * 8));
      bf16x16 b1 = mk_frag(*(const bf16x8*)(k1p + kf * 32 + hlf * 8),
                           *(const bf16x8*)(k1p + kf * 32 + 16 + hlf * 8));
      s0 = wmma_bf16(qF[kf], b0, s0);
      s1 = wmma_bf16(qF[kf], b1, s1);
    }
    float corr[8];
#pragma unroll
    for (int r = 0; r < 8; ++r) {
      float a = s0[r] * scale, c = s1[r] * scale;
      float v = fmaxf(a, c);
#pragma unroll
      for (int off = 8; off; off >>= 1) v = fmaxf(v, __shfl_xor(v, off, 32));
      const float mn = fmaxf(mrow[r], v);
      corr[r] = __expf(mrow[r] - mn);
      const float p0 = __expf(a - mn);
      const float p1 = __expf(c - mn);
      s0[r] = p0; s1[r] = p1;
      float ps = p0 + p1;
#pragma unroll
      for (int off = 8; off; off >>= 1) ps += __shfl_xor(ps, off, 32);
      lrow[r] = lrow[r] * corr[r] + ps;
      mrow[r] = mn;
    }
#pragma unroll
    for (int nt = 0; nt < 8; ++nt)
#pragma unroll
      for (int r = 0; r < 8; ++r) o[nt][r] *= corr[r];
    // P tile: C-layout -> row-major 16x32 in LDS -> reread as A-operand frag
#pragma unroll
    for (int r = 0; r < 8; ++r) {
      const int row = r + hlf * 8;
      pw[row * 40 + ln]      = (bf16)s0[r];
      pw[row * 40 + 16 + ln] = (bf16)s1[r];
    }
    __syncthreads();  // all waves have identical trip counts
    bf16x16 pF = mk_frag(*(const bf16x8*)(pw + ln * 40 + hlf * 8),
                         *(const bf16x8*)(pw + ln * 40 + 16 + hlf * 8));
#pragma unroll
    for (int nt = 0; nt < 8; ++nt) {
      const bf16* vp = vHead + (size_t)(nt * 16 + ln) * T_ + kb * 32;
      bf16x16 bv = mk_frag(*(const bf16x8*)(vp + hlf * 8),
                           *(const bf16x8*)(vp + 16 + hlf * 8));
      o[nt] = wmma_bf16(pF, bv, o[nt]);
    }
  }
  float inv[8];
#pragma unroll
  for (int r = 0; r < 8; ++r) inv[r] = 1.f / lrow[r];
#pragma unroll
  for (int nt = 0; nt < 8; ++nt)
#pragma unroll
    for (int r = 0; r < 8; ++r) {
      const int tok = q0 + r + hlf * 8;
      Oa[((size_t)b * T_ + tok) * (NH_ * VD_) + h * VD_ + nt * 16 + ln] =
          (bf16)(o[nt][r] * inv[r]);
    }
}

// ---------------------------------------------------------------------------
// Prep / epilogue kernels
// ---------------------------------------------------------------------------
__global__ __launch_bounds__(256) void mla_cast_bf16(const float* __restrict__ in,
                                                     bf16* __restrict__ out, int n) {
  const int i = blockIdx.x * 256 + threadIdx.x;
  if (i < n) out[i] = (bf16)in[i];
}

// W[K][N] f32 -> Wt[N][K] bf16
__global__ __launch_bounds__(256) void mla_transpose_cast(
    const float* __restrict__ W, bf16* __restrict__ Wt, int K, int N) {
  __shared__ float tile[32][33];
  const int tx = threadIdx.x & 31, ty = threadIdx.x >> 5;
  const int k0 = blockIdx.y * 32, n0 = blockIdx.x * 32;
#pragma unroll
  for (int i = 0; i < 32; i += 8)
    tile[ty + i][tx] = W[(size_t)(k0 + ty + i) * N + n0 + tx];
  __syncthreads();
#pragma unroll
  for (int i = 0; i < 32; i += 8)
    Wt[(size_t)(n0 + ty + i) * K + k0 + tx] = (bf16)tile[tx][ty + i];
}

__global__ __launch_bounds__(256) void mla_rmsnorm_cast(
    const float* __restrict__ in, const float* __restrict__ w,
    bf16* __restrict__ out, int width) {
  const int row  = blockIdx.x;
  const float* x = in + (size_t)row * width;
  float ss = 0.f;
  for (int i = threadIdx.x; i < width; i += 256) { const float v = x[i]; ss += v * v; }
  const float inv = rsqrtf(block_sum(ss) / (float)width + EPS_);
  for (int i = threadIdx.x; i < width; i += 256)
    out[(size_t)row * width + i] = (bf16)(x[i] * inv * w[i]);
}

// q f32 [ROWS, NH*HD] -> qfull bf16 [B,NH,T,HD] with RoPE on dims [NOPE,HD)
__global__ __launch_bounds__(256) void mla_q_post(const float* __restrict__ q,
                                                  bf16* __restrict__ qfull) {
  const int row = blockIdx.x;
  const int b = row / T_, t = row % T_;
  const float* qr = q + (size_t)row * (NH_ * HD_);
  for (int idx = threadIdx.x; idx < NH_ * HD_; idx += 256) {
    const int h = idx / HD_, d = idx % HD_;
    float v = qr[idx];
    if (d >= NOPE_) {
      const int j = d - NOPE_, jm = j & 31;
      const float f = __expf(-(float)jm * LOG_THETA_OVER_HALF);
      float s, c;
      __sincosf((float)t * f, &s, &c);
      const float other = (j < 32) ? -qr[h * HD_ + NOPE_ + j + 32]
                                   :  qr[h * HD_ + NOPE_ + j - 32];
      v = v * c + other * s;
    }
    qfull[(((size_t)b * NH_ + h) * T_ + t) * HD_ + d] = (bf16)v;
  }
}

// kv_raw f32 [ROWS, KVR+NKV*ROPE]: rmsnorm->latent bf16; RoPE k_rope into kfull
__global__ __launch_bounds__(256) void mla_kv_a_post(
    const float* __restrict__ kvraw, const float* __restrict__ w,
    bf16* __restrict__ latent, bf16* __restrict__ kfull) {
  const int row = blockIdx.x;
  const int b = row / T_, t = row % T_;
  const float* x = kvraw + (size_t)row * (KVR_ + NKV_ * ROPE_);
  float ss = 0.f;
  for (int i = threadIdx.x; i < KVR_; i += 256) { const float v = x[i]; ss += v * v; }
  const float inv = rsqrtf(block_sum(ss) / (float)KVR_ + EPS_);
  for (int i = threadIdx.x; i < KVR_; i += 256)
    latent[(size_t)row * KVR_ + i] = (bf16)(x[i] * inv * w[i]);
  for (int i = threadIdx.x; i < NKV_ * ROPE_; i += 256) {
    const int kvh = i >> 6, j = i & 63, jm = j & 31;
    const float f = __expf(-(float)jm * LOG_THETA_OVER_HALF);
    float s, c;
    __sincosf((float)t * f, &s, &c);
    const float v = x[KVR_ + i];
    const float other = (j < 32) ? -x[KVR_ + kvh * ROPE_ + j + 32]
                                 :  x[KVR_ + kvh * ROPE_ + j - 32];
    kfull[(((size_t)b * NKV_ + kvh) * T_ + t) * HD_ + NOPE_ + j] =
        (bf16)(v * c + other * s);
  }
}

// kv f32 [ROWS, NKV*(NOPE+VD)] -> k_nope into kfull, V transposed into Vt
__global__ __launch_bounds__(256) void mla_kv_post(const float* __restrict__ kv,
                                                   bf16* __restrict__ kfull,
                                                   bf16* __restrict__ vt) {
  const size_t gid = (size_t)blockIdx.x * 256 + threadIdx.x;
  const size_t total = (size_t)ROWS_ * (NKV_ * (NOPE_ + VD_));
  if (gid >= total) return;
  const int row  = (int)(gid / (NKV_ * (NOPE_ + VD_)));
  const int cidx = (int)(gid % (NKV_ * (NOPE_ + VD_)));
  const int kvh = cidx >> 8, d = cidx & 255;
  const int b = row / T_, t = row % T_;
  const float v = kv[gid];
  if (d < NOPE_)
    kfull[(((size_t)b * NKV_ + kvh) * T_ + t) * HD_ + d] = (bf16)v;
  else
    vt[(((size_t)b * NKV_ + kvh) * VD_ + (d - NOPE_)) * T_ + t] = (bf16)v;
}

// ---------------------------------------------------------------------------
extern "C" void kernel_launch(void* const* d_in, const int* in_sizes, int n_in,
                              void* d_out, int out_size, void* d_ws, size_t ws_size,
                              hipStream_t stream) {
  (void)in_sizes; (void)n_in; (void)out_size; (void)ws_size;
  const float* hs      = (const float*)d_in[0];
  const float* q_a_w   = (const float*)d_in[1];
  const float* q_a_ln  = (const float*)d_in[2];
  const float* q_b_w   = (const float*)d_in[3];
  const float* kv_a_w  = (const float*)d_in[4];
  const float* kv_a_ln = (const float*)d_in[5];
  const float* kv_b_w  = (const float*)d_in[6];
  const float* o_w     = (const float*)d_in[7];
  float* out = (float*)d_out;

  char* p = (char*)d_ws;
  auto take = [&](size_t elems, size_t esz) -> void* {
    void* r = (void*)p;
    p += (elems * esz + 255) & ~(size_t)255;
    return r;
  };
  bf16*  Xbf     = (bf16*)take((size_t)ROWS_ * HID_, 2);
  bf16*  qa_wT   = (bf16*)take((size_t)QR_ * HID_, 2);
  bf16*  qb_wT   = (bf16*)take((size_t)NH_ * HD_ * QR_, 2);
  bf16*  kva_wT  = (bf16*)take((size_t)(KVR_ + NKV_ * ROPE_) * HID_, 2);
  bf16*  kvb_wT  = (bf16*)take((size_t)NKV_ * (NOPE_ + VD_) * KVR_, 2);
  bf16*  o_wT    = (bf16*)take((size_t)HID_ * NH_ * VD_, 2);
  float* qa      = (float*)take((size_t)ROWS_ * QR_, 4);
  bf16*  qa_ln_b = (bf16*)take((size_t)ROWS_ * QR_, 2);
  float* qf32    = (float*)take((size_t)ROWS_ * NH_ * HD_, 4);
  bf16*  qfull   = (bf16*)take((size_t)B_ * NH_ * T_ * HD_, 2);
  float* kvraw   = (float*)take((size_t)ROWS_ * (KVR_ + NKV_ * ROPE_), 4);
  bf16*  latent  = (bf16*)take((size_t)ROWS_ * KVR_, 2);
  float* kvf32   = (float*)take((size_t)ROWS_ * NKV_ * (NOPE_ + VD_), 4);
  bf16*  kfull   = (bf16*)take((size_t)B_ * NKV_ * T_ * HD_, 2);
  bf16*  vt      = (bf16*)take((size_t)B_ * NKV_ * VD_ * T_, 2);
  bf16*  attn    = (bf16*)take((size_t)ROWS_ * NH_ * VD_, 2);

  const int n_x = ROWS_ * HID_;
  mla_cast_bf16<<<(n_x + 255) / 256, 256, 0, stream>>>(hs, Xbf, n_x);
  mla_transpose_cast<<<dim3(QR_ / 32, HID_ / 32), 256, 0, stream>>>(
      q_a_w, qa_wT, HID_, QR_);
  mla_transpose_cast<<<dim3(NH_ * HD_ / 32, QR_ / 32), 256, 0, stream>>>(
      q_b_w, qb_wT, QR_, NH_ * HD_);
  mla_transpose_cast<<<dim3((KVR_ + NKV_ * ROPE_) / 32, HID_ / 32), 256, 0, stream>>>(
      kv_a_w, kva_wT, HID_, KVR_ + NKV_ * ROPE_);
  mla_transpose_cast<<<dim3(NKV_ * (NOPE_ + VD_) / 32, KVR_ / 32), 256, 0, stream>>>(
      kv_b_w, kvb_wT, KVR_, NKV_ * (NOPE_ + VD_));
  mla_transpose_cast<<<dim3(HID_ / 32, (NH_ * VD_) / 32), 256, 0, stream>>>(
      o_w, o_wT, NH_ * VD_, HID_);

  mla_gemm_bf16<<<dim3(QR_ / 128, ROWS_ / 128), 256, 0, stream>>>(
      Xbf, qa_wT, qa, ROWS_, QR_, HID_);
  mla_rmsnorm_cast<<<ROWS_, 256, 0, stream>>>(qa, q_a_ln, qa_ln_b, QR_);
  mla_gemm_bf16<<<dim3(NH_ * HD_ / 128, ROWS_ / 128), 256, 0, stream>>>(
      qa_ln_b, qb_wT, qf32, ROWS_, NH_ * HD_, QR_);
  mla_q_post<<<ROWS_, 256, 0, stream>>>(qf32, qfull);

  mla_gemm_bf16<<<dim3((KVR_ + NKV_ * ROPE_) / 128, ROWS_ / 128), 256, 0, stream>>>(
      Xbf, kva_wT, kvraw, ROWS_, KVR_ + NKV_ * ROPE_, HID_);
  mla_kv_a_post<<<ROWS_, 256, 0, stream>>>(kvraw, kv_a_ln, latent, kfull);
  mla_gemm_bf16<<<dim3(NKV_ * (NOPE_ + VD_) / 128, ROWS_ / 128), 256, 0, stream>>>(
      latent, kvb_wT, kvf32, ROWS_, NKV_ * (NOPE_ + VD_), KVR_);
  const size_t nkv = (size_t)ROWS_ * NKV_ * (NOPE_ + VD_);
  mla_kv_post<<<(unsigned)((nkv + 255) / 256), 256, 0, stream>>>(kvf32, kfull, vt);

  mla_attention<<<dim3(T_ / 128, NH_, B_), 256, 0, stream>>>(qfull, kfull, vt, attn);

  mla_gemm_bf16<<<dim3(HID_ / 128, ROWS_ / 128), 256, 0, stream>>>(
      attn, o_wT, out, ROWS_, HID_, HID_);
}
